// Fuse_2310692405265
// MI455X (gfx1250) — matrix operand order repaired and checked by reading, not compiled, not run on hardware
//
#include <hip/hip_runtime.h>
#include <hip/hip_bf16.h>

// ---------------- types ----------------
typedef __attribute__((ext_vector_type(16))) _Float16 v16h;
typedef __attribute__((ext_vector_type(8)))  _Float16 v8h;
typedef __attribute__((ext_vector_type(8)))  float    v8f;

#define H_DIM   768
#define B_DIM   4
#define L_DIM   512
#define Q_DIM   64
#define NEG_INF (-1e25f)
#define L2E     1.4426950408889634f

// ---------------- elementwise f32 -> f16 ----------------
__global__ void cvt_f32_to_f16(const float* __restrict__ src,
                               _Float16* __restrict__ dst, int n) {
    int i = blockIdx.x * blockDim.x + threadIdx.x;
    if (i < n) dst[i] = (_Float16)src[i];
}

// Split W_w (H x 2H, row major) into W_bert (H x H) and W_query (H x H), f16
__global__ void split_w_f16(const float* __restrict__ Ww,
                            _Float16* __restrict__ Wb,
                            _Float16* __restrict__ Wq, int n /* = H*H */) {
    int i = blockIdx.x * blockDim.x + threadIdx.x;
    if (i >= n) return;
    int k = i / H_DIM;
    int h = i - k * H_DIM;
    Wb[i] = (_Float16)Ww[k * (2 * H_DIM) + h];
    Wq[i] = (_Float16)Ww[k * (2 * H_DIM) + H_DIM + h];
}

// ---------------- WMMA NT GEMM: C[m,n] = sum_k A[m,k] * W[n,k] ----------------
// A: M x K row-major f16. W: N x K row-major f16. C: M x N row-major f32.
// One wave per 32x64 C macro-tile: 2 A-frags x 4 B-frags -> 8 WMMAs / k-step
// (12 b128 loads per 8 WMMAs instead of 32).
// grid = (N/64, M/32), block = 32.
__global__ __launch_bounds__(32)
void gemm_nt_wmma(const _Float16* __restrict__ A,
                  const _Float16* __restrict__ W,
                  float* __restrict__ C,
                  int M, int N, int K) {
    const int n0   = blockIdx.x * 64;
    const int m0   = blockIdx.y * 32;
    const int lane = threadIdx.x;          // 0..31, all active
    const int r    = lane & 15;
    const int half = lane >> 4;            // 0 or 1

    // f16 A-frag (16x32): lane holds row r; 16 contiguous K at kbA.
    const int kbA = half * 16;
    // f16 B-frag (32x16): lane holds column r; 16 contiguous K at kbB.
    const int kbB = half * 16;

    const _Float16* arow0 = A + (size_t)(m0 + r) * K;
    const _Float16* arow1 = arow0 + (size_t)16 * K;
    const _Float16* wrow[4];
    #pragma unroll
    for (int j = 0; j < 4; ++j)
        wrow[j] = W + (size_t)(n0 + j * 16 + r) * K;

    v8f acc[2][4];
    #pragma unroll
    for (int mi = 0; mi < 2; ++mi)
        #pragma unroll
        for (int j = 0; j < 4; ++j)
            acc[mi][j] = (v8f){};

    for (int k0 = 0; k0 < K; k0 += 32) {
        v8h a00 = *(const v8h*)(arow0 + k0 + kbA);
        v8h a01 = *(const v8h*)(arow0 + k0 + kbA + 8);
        v8h a10 = *(const v8h*)(arow1 + k0 + kbA);
        v8h a11 = *(const v8h*)(arow1 + k0 + kbA + 8);
        v16h a0 = __builtin_shufflevector(a00, a01, 0,1,2,3,4,5,6,7,
                                                    8,9,10,11,12,13,14,15);
        v16h a1 = __builtin_shufflevector(a10, a11, 0,1,2,3,4,5,6,7,
                                                    8,9,10,11,12,13,14,15);
        #pragma unroll
        for (int j = 0; j < 4; ++j) {
            v16h b = *(const v16h*)(wrow[j] + k0 + kbB);
            acc[0][j] = __builtin_amdgcn_wmma_f32_16x16x32_f16(
                            false, a0, false, b, (short)0, acc[0][j],
                            false, false);
            acc[1][j] = __builtin_amdgcn_wmma_f32_16x16x32_f16(
                            false, a1, false, b, (short)0, acc[1][j],
                            false, false);
        }
    }

    // f32 C/D layout: VGPR v -> row M = v (lanes 0-15) or v+8 (lanes 16-31),
    // column N = lane & 15.
    #pragma unroll
    for (int mi = 0; mi < 2; ++mi) {
        #pragma unroll
        for (int j = 0; j < 4; ++j) {
            float* crow = C + (size_t)(m0 + mi * 16 + half * 8) * N
                            + (n0 + j * 16 + r);
            #pragma unroll
            for (int v = 0; v < 8; ++v)
                crow[(size_t)v * N] = acc[mi][j][v];
        }
    }
}

// ---------------- fused tanh-reduce + masked softmax ----------------
__device__ __forceinline__ float tanh_fast(float x) {
#if __has_builtin(__builtin_amdgcn_tanhf)
    return __builtin_amdgcn_tanhf(x);     // v_tanh_f32 (1 trans op)
#else
    // tanh(x) = 1 - 2/(exp2(2*log2e*x)+1); saturates correctly at +-1.
    float t = __builtin_amdgcn_exp2f(x * (2.0f * L2E));
    return 1.0f - 2.0f * __builtin_amdgcn_rcpf(t + 1.0f);
#endif
}

// One block per (b,q). 256 threads = 8 waves; wave w owns rows l = w, w+8,...
// qp+W_b and v_w slices are register-resident per lane (fixed lane->h map),
// so the hot loop is global b128 loads + trans/FMA only.
__global__ __launch_bounds__(256)
void attn_score_softmax(const float* __restrict__ bp,   // (B*L, H)
                        const float* __restrict__ qp,   // (B*Q, H)
                        const float* __restrict__ Wb,   // (H,)
                        const float* __restrict__ vw,   // (H,)
                        const float* __restrict__ vb,   // (1,)
                        const unsigned char* __restrict__ mask, // (B, L) bool
                        float* __restrict__ out) {      // (B*Q, L)
    __shared__ float xs[L_DIM];
    __shared__ float red[256];

    const int bq = blockIdx.x;          // 0..255
    const int b  = bq >> 6;             // / Q_DIM
    const int tid  = threadIdx.x;
    const int wave = tid >> 5;
    const int lane = tid & 31;

    // Register-cache qw = qp[b,q,hslice] + W_b[hslice] and vw[hslice].
    float4 q4[H_DIM / 128];
    float4 v4[H_DIM / 128];
    const float* qrow = qp + (size_t)bq * H_DIM;
    #pragma unroll
    for (int it = 0; it < H_DIM / 128; ++it) {
        const int idx = it * 128 + lane * 4;
        float4 qq = *(const float4*)(qrow + idx);
        float4 wb = *(const float4*)(Wb + idx);
        q4[it].x = qq.x + wb.x;  q4[it].y = qq.y + wb.y;
        q4[it].z = qq.z + wb.z;  q4[it].w = qq.w + wb.w;
        v4[it]   = *(const float4*)(vw + idx);
    }

    const float vbias = vb[0];
    const unsigned char* mrow = mask + (size_t)b * L_DIM;

    #pragma unroll 2
    for (int l = wave; l < L_DIM; l += 8) {
        const float* bpr = bp + ((size_t)b * L_DIM + l) * H_DIM + lane * 4;
        float ax = 0.f, ay = 0.f, az = 0.f, aw = 0.f;
        #pragma unroll
        for (int it = 0; it < H_DIM / 128; ++it) {   // 6 b128 loads per row
            float4 x4 = *(const float4*)(bpr + it * 128);
            ax += tanh_fast(x4.x + q4[it].x) * v4[it].x;
            ay += tanh_fast(x4.y + q4[it].y) * v4[it].y;
            az += tanh_fast(x4.z + q4[it].z) * v4[it].z;
            aw += tanh_fast(x4.w + q4[it].w) * v4[it].w;
        }
        float a = (ax + ay) + (az + aw);
        #pragma unroll
        for (int off = 16; off > 0; off >>= 1)
            a += __shfl_xor(a, off, 32);
        if (lane == 0) {
            float x = a + vbias;
            if (!mrow[l]) x = NEG_INF;
            xs[l] = x;
        }
    }
    __syncthreads();

    // Masked softmax over L=512 (2 elements per thread).
    float x0 = xs[tid];
    float x1 = xs[tid + 256];
    red[tid] = fmaxf(x0, x1);
    __syncthreads();
    #pragma unroll
    for (int s = 128; s > 0; s >>= 1) {
        if (tid < s) red[tid] = fmaxf(red[tid], red[tid + s]);
        __syncthreads();
    }
    const float M = red[0];
    __syncthreads();

    float e0 = __builtin_amdgcn_exp2f((x0 - M) * L2E);
    float e1 = __builtin_amdgcn_exp2f((x1 - M) * L2E);
    red[tid] = e0 + e1;
    __syncthreads();
    #pragma unroll
    for (int s = 128; s > 0; s >>= 1) {
        if (tid < s) red[tid] += red[tid + s];
        __syncthreads();
    }
    const float inv = __builtin_amdgcn_rcpf(red[0]);

    float* orow = out + (size_t)bq * L_DIM;
    orow[tid]       = e0 * inv;
    orow[tid + 256] = e1 * inv;
}

// ---------------- launch ----------------
extern "C" void kernel_launch(void* const* d_in, const int* in_sizes, int n_in,
                              void* d_out, int out_size, void* d_ws, size_t ws_size,
                              hipStream_t stream) {
    const float* bert  = (const float*)d_in[0];   // (4, 512, 768)
    const float* query = (const float*)d_in[1];   // (4, 64, 768)
    const unsigned char* mask = (const unsigned char*)d_in[2]; // (4,512) bool
    const float* W_w   = (const float*)d_in[3];   // (768, 1536)
    const float* W_b   = (const float*)d_in[4];   // (768,)
    const float* v_w   = (const float*)d_in[5];   // (1, 768)
    const float* v_b   = (const float*)d_in[6];   // (1,)
    float* out = (float*)d_out;                   // (4, 64, 512)

    const int MB = B_DIM * L_DIM;   // 2048 rows of bert
    const int MQ = B_DIM * Q_DIM;   // 256 rows of query
    const int HH = H_DIM * H_DIM;   // 589824

    // Workspace layout (all sizes multiples of 256 bytes).
    char* ws = (char*)d_ws;
    _Float16* bert_h  = (_Float16*)(ws);                                    // 3 MB
    _Float16* query_h = (_Float16*)(ws + (size_t)MB * H_DIM * 2);           // 384 KB
    _Float16* Wb_h    = (_Float16*)(ws + (size_t)(MB + MQ) * H_DIM * 2);    // 1.125 MB
    _Float16* Wq_h    = Wb_h + HH;
    float*    bp      = (float*)((char*)(Wq_h + HH));                       // 6 MB
    float*    qp      = bp + (size_t)MB * H_DIM;                            // 768 KB

    // 1) f32 -> f16 conversions
    {
        int n = MB * H_DIM;
        cvt_f32_to_f16<<<(n + 255) / 256, 256, 0, stream>>>(bert, bert_h, n);
    }
    {
        int n = MQ * H_DIM;
        cvt_f32_to_f16<<<(n + 255) / 256, 256, 0, stream>>>(query, query_h, n);
    }
    split_w_f16<<<(HH + 255) / 256, 256, 0, stream>>>(W_w, Wb_h, Wq_h, HH);

    // 2) WMMA GEMMs: bp = bert @ W_bert^T, qp = query @ W_query^T
    gemm_nt_wmma<<<dim3(H_DIM / 64, MB / 32), 32, 0, stream>>>(
        bert_h, Wb_h, bp, MB, H_DIM, H_DIM);
    gemm_nt_wmma<<<dim3(H_DIM / 64, MQ / 32), 32, 0, stream>>>(
        query_h, Wq_h, qp, MQ, H_DIM, H_DIM);

    // 3) fused tanh reduction + masked softmax
    attn_score_softmax<<<B_DIM * Q_DIM, 256, 0, stream>>>(
        bp, qp, W_b, v_w, v_b, mask, out);
}
